// MultiHeadedAttention_31671088841042
// MI455X (gfx1250) — matrix operand order repaired and checked by reading, not compiled, not run on hardware
//
#include <hip/hip_runtime.h>
#include <hip/hip_bf16.h>
#include <cstdint>

#define B_SZ   4
#define S_LEN  2048
#define D_DIM  1024
#define HEADS  16
#define DK     64
#define M_ROWS (B_SZ * S_LEN)   // 8192

typedef __attribute__((ext_vector_type(16))) __bf16 bf16x16;
typedef __attribute__((ext_vector_type(2)))  __bf16 bf16x2;
typedef __attribute__((ext_vector_type(8)))  float  f32x8;
typedef __attribute__((ext_vector_type(4)))  unsigned int u32x4;

union ABFrag { bf16x16 bf; u32x4 q[2]; };
union H8     { u32x4 v; unsigned short s[8]; };

// ---- fp32 -> bf16 packing: hardware v_cvt_pk_bf16_f32 when available -------
#if __has_builtin(__builtin_amdgcn_cvt_pk_bf16_f32)
static __device__ __forceinline__ unsigned int pk_bf16_rne(float lo, float hi) {
  bf16x2 p = __builtin_amdgcn_cvt_pk_bf16_f32(lo, hi);
  return __builtin_bit_cast(unsigned int, p);
}
#else
static __device__ __forceinline__ unsigned int pk_bf16_rne(float lo, float hi) {
  unsigned int a = __float_as_uint(lo), b = __float_as_uint(hi);
  a += 0x7FFFu + ((a >> 16) & 1u);
  b += 0x7FFFu + ((b >> 16) & 1u);
  return (a >> 16) | (b & 0xFFFF0000u);
}
#endif
static __device__ __forceinline__ unsigned short f2bf(float v) {
  return (unsigned short)(pk_bf16_rne(v, 0.0f) & 0xFFFFu);
}

static __device__ __forceinline__ f32x8 wmma_bf16(const ABFrag& a, const ABFrag& b, f32x8 c) {
  return __builtin_amdgcn_wmma_f32_16x16x32_bf16(false, a.bf, false, b.bf,
                                                 (short)0, c, false, false);
}

// ---------------------------------------------------------------------------
// fp32 -> bf16 elementwise, 8 floats / thread (memory bound)
// ---------------------------------------------------------------------------
__global__ void cvt_f32_to_bf16(const float* __restrict__ src,
                                unsigned short* __restrict__ dst, int n8) {
  int i = blockIdx.x * blockDim.x + threadIdx.x;
  if (i < n8) {
    float4 a = ((const float4*)src)[2 * i];
    float4 b = ((const float4*)src)[2 * i + 1];
    u32x4 p;
    p[0] = pk_bf16_rne(a.x, a.y);
    p[1] = pk_bf16_rne(a.z, a.w);
    p[2] = pk_bf16_rne(b.x, b.y);
    p[3] = pk_bf16_rne(b.z, b.w);
    ((u32x4*)dst)[i] = p;
  }
}

// ---------------------------------------------------------------------------
// Dense GEMM: C = A(bf16) @ W(bf16)^T
// block = 128 threads (4 waves); block tile 128(M) x 64(N); wave tile 32 x 64
// MODE 0: store bf16 into head-split [B,H,S,DK]   (Q/K/V projections)
// MODE 1: store fp32 row-major [M,N]              (output projection)
// ---------------------------------------------------------------------------
template <int MODE>
__global__ __launch_bounds__(128)
void gemm_kernel(const unsigned short* __restrict__ Ab,
                 const unsigned short* __restrict__ Wb,
                 void* __restrict__ OutV) {
  const int lane = threadIdx.x & 31;
  const int wave = threadIdx.x >> 5;
  const int r = lane & 15;
  const int h = lane >> 4;
  const int mbase = blockIdx.x * 128 + wave * 32;
  const int nbase = blockIdx.y * 64;

  f32x8 acc[2][4];
#pragma unroll
  for (int u = 0; u < 2; ++u)
#pragma unroll
    for (int t = 0; t < 4; ++t)
#pragma unroll
      for (int e = 0; e < 8; ++e) acc[u][t][e] = 0.0f;

  const unsigned short* arow0 = Ab + (size_t)(mbase + r) * D_DIM;
  const unsigned short* arow1 = arow0 + (size_t)16 * D_DIM;

  for (int kb = 0; kb < D_DIM; kb += 32) {
    ABFrag a0, a1;
    a0.q[0] = *(const u32x4*)(arow0 + kb + 8 * h);
    a0.q[1] = *(const u32x4*)(arow0 + kb + 16 + 8 * h);
    a1.q[0] = *(const u32x4*)(arow1 + kb + 8 * h);
    a1.q[1] = *(const u32x4*)(arow1 + kb + 16 + 8 * h);
#pragma unroll
    for (int t = 0; t < 4; ++t) {
      // B-frag column n = nbase+16t+r == row n of W (contiguous over k)
      const unsigned short* brow =
          Wb + (size_t)(nbase + 16 * t + r) * D_DIM + kb + 8 * h;
      ABFrag b;
      b.q[0] = *(const u32x4*)(brow);
      b.q[1] = *(const u32x4*)(brow + 16);
      acc[0][t] = wmma_bf16(a0, b, acc[0][t]);
      acc[1][t] = wmma_bf16(a1, b, acc[1][t]);
    }
  }

#pragma unroll
  for (int u = 0; u < 2; ++u) {
#pragma unroll
    for (int t = 0; t < 4; ++t) {
#pragma unroll
      for (int i = 0; i < 8; ++i) {
        int g = mbase + u * 16 + 8 * h + i;   // global row
        int j = nbase + 16 * t + r;           // global col
        if (MODE == 0) {
          int bb = g >> 11, s = g & 2047;
          int hd = j >> 6, d = j & 63;
          ((unsigned short*)OutV)[(((size_t)bb * HEADS + hd) * S_LEN + s) * DK + d] =
              f2bf(acc[u][t][i]);
        } else {
          ((float*)OutV)[(size_t)g * D_DIM + j] = acc[u][t][i];
        }
      }
    }
  }
}

// ---------------------------------------------------------------------------
// Flash attention: one workgroup = (b,h) x 64 query rows; 4 waves x 16 rows.
// Streams key tiles of 64, online softmax, mask applied as in reference.
// ---------------------------------------------------------------------------
__global__ __launch_bounds__(128)
void flash_attn_kernel(const unsigned short* __restrict__ Qp,
                       const unsigned short* __restrict__ Kp,
                       const unsigned short* __restrict__ Vp,
                       const int* __restrict__ mask,
                       unsigned short* __restrict__ AttnOut) {
  __shared__ __align__(16) unsigned short vT[64 * 80];      // V^T tile [d][key]
  __shared__ __align__(16) unsigned short pS[4 * 16 * 72];  // P per wave [16][72]

  const int lane = threadIdx.x & 31;
  const int wave = threadIdx.x >> 5;
  const int r = lane & 15;
  const int h = lane >> 4;
  const int bh   = blockIdx.y;
  const int b    = bh >> 4;
  const int head = bh & 15;
  const int qrow = blockIdx.x * 64 + wave * 16;

  const size_t headoff = (size_t)bh * S_LEN * DK;
  const unsigned short* qptr = Qp + headoff + (size_t)(qrow + r) * DK;

  ABFrag qa[2];
  qa[0].q[0] = *(const u32x4*)(qptr + 8 * h);
  qa[0].q[1] = *(const u32x4*)(qptr + 16 + 8 * h);
  qa[1].q[0] = *(const u32x4*)(qptr + 32 + 8 * h);
  qa[1].q[1] = *(const u32x4*)(qptr + 48 + 8 * h);

  float m_i[8], l_i[8];
#pragma unroll
  for (int i = 0; i < 8; ++i) { m_i[i] = -3.0e38f; l_i[i] = 0.0f; }
  f32x8 o[4];
#pragma unroll
  for (int t = 0; t < 4; ++t)
#pragma unroll
    for (int e = 0; e < 8; ++e) o[t][e] = 0.0f;

  const float scale = 0.125f;  // 1/sqrt(64)
  const int* mrow = mask + ((size_t)b * S_LEN + qrow) * S_LEN;
  unsigned short* pw = pS + wave * 16 * 72;

  for (int kb = 0; kb < S_LEN; kb += 64) {
    __syncthreads();
    // cooperative transpose of V tile (64 keys x 64 d) into vT[d][key]
    for (int c = threadIdx.x; c < 512; c += 128) {
      int key = c >> 3;
      int d0  = (c & 7) * 8;
      H8 u;
      u.v = *(const u32x4*)(Vp + headoff + (size_t)(kb + key) * DK + d0);
#pragma unroll
      for (int j = 0; j < 8; ++j) vT[(d0 + j) * 80 + key] = u.s[j];
    }
    __syncthreads();

    // S tile = Q * K^T : 4 subtiles of 16x16, each 2 WMMAs over dk
    f32x8 st[4];
#pragma unroll
    for (int t = 0; t < 4; ++t) {
#pragma unroll
      for (int e = 0; e < 8; ++e) st[t][e] = 0.0f;
      const unsigned short* krow =
          Kp + headoff + (size_t)(kb + 16 * t + r) * DK + 8 * h;
      ABFrag b0, b1;
      b0.q[0] = *(const u32x4*)(krow);
      b0.q[1] = *(const u32x4*)(krow + 16);
      b1.q[0] = *(const u32x4*)(krow + 32);
      b1.q[1] = *(const u32x4*)(krow + 48);
      st[t] = wmma_bf16(qa[0], b0, st[t]);
      st[t] = wmma_bf16(qa[1], b1, st[t]);
    }

    if (kb + 64 < S_LEN)
      __builtin_prefetch(Kp + headoff + (size_t)(kb + 64 + lane * 2) * DK, 0, 1);

    // mask + scale, per-row max (rows striped across i for this lane's half)
    float rowmax[8];
#pragma unroll
    for (int i = 0; i < 8; ++i) rowmax[i] = -3.0e38f;
#pragma unroll
    for (int t = 0; t < 4; ++t)
#pragma unroll
      for (int i = 0; i < 8; ++i) {
        int mv  = mrow[(8 * h + i) * S_LEN + (kb + 16 * t + r)];
        float s = (mv != 0) ? st[t][i] * scale : -1.0e9f;
        st[t][i] = s;
        rowmax[i] = fmaxf(rowmax[i], s);
      }
#pragma unroll
    for (int off = 1; off < 16; off <<= 1)
#pragma unroll
      for (int i = 0; i < 8; ++i)
        rowmax[i] = fmaxf(rowmax[i], __shfl_xor(rowmax[i], off, 16));

    float m_new[8], alpha[8], rowsum[8];
#pragma unroll
    for (int i = 0; i < 8; ++i) {
      m_new[i]  = fmaxf(m_i[i], rowmax[i]);
      alpha[i]  = __expf(m_i[i] - m_new[i]);
      rowsum[i] = 0.0f;
    }
#pragma unroll
    for (int t = 0; t < 4; ++t)
#pragma unroll
      for (int i = 0; i < 8; ++i) {
        float s = st[t][i];
        float p = (s > -5.0e8f) ? __expf(s - m_new[i]) : 0.0f;  // masked -> 0
        st[t][i] = p;
        rowsum[i] += p;
      }
#pragma unroll
    for (int off = 1; off < 16; off <<= 1)
#pragma unroll
      for (int i = 0; i < 8; ++i)
        rowsum[i] += __shfl_xor(rowsum[i], off, 16);
#pragma unroll
    for (int i = 0; i < 8; ++i) {
      l_i[i] = l_i[i] * alpha[i] + rowsum[i];
      m_i[i] = m_new[i];
    }
#pragma unroll
    for (int t = 0; t < 4; ++t)
#pragma unroll
      for (int i = 0; i < 8; ++i) o[t][i] *= alpha[i];

    // P (C-layout) -> LDS -> A-layout frags
#pragma unroll
    for (int t = 0; t < 4; ++t)
#pragma unroll
      for (int i = 0; i < 8; ++i)
        pw[(8 * h + i) * 72 + 16 * t + r] = f2bf(st[t][i]);

    asm volatile("s_wait_dscnt 0" ::: "memory");  // wave-local DS RAW

    ABFrag p0, p1;
    const unsigned short* prow = pw + r * 72;
    p0.q[0] = *(const u32x4*)(prow + 8 * h);
    p0.q[1] = *(const u32x4*)(prow + 16 + 8 * h);
    p1.q[0] = *(const u32x4*)(prow + 32 + 8 * h);
    p1.q[1] = *(const u32x4*)(prow + 48 + 8 * h);

    // O += P * V : B-operand columns are rows of vT (contiguous)
#pragma unroll
    for (int t = 0; t < 4; ++t) {
      const unsigned short* vrow = vT + (16 * t + r) * 80;
      ABFrag b0, b1;
      b0.q[0] = *(const u32x4*)(vrow + 8 * h);
      b0.q[1] = *(const u32x4*)(vrow + 16 + 8 * h);
      b1.q[0] = *(const u32x4*)(vrow + 32 + 8 * h);
      b1.q[1] = *(const u32x4*)(vrow + 48 + 8 * h);
      o[t] = wmma_bf16(p0, b0, o[t]);
      o[t] = wmma_bf16(p1, b1, o[t]);
    }
  }

  // normalize and store bf16 into concat-head layout [B,S,D], col = head*64+d
#pragma unroll
  for (int i = 0; i < 8; ++i) {
    float inv = (l_i[i] > 0.0f) ? 1.0f / l_i[i] : 0.0f;
    int q = qrow + 8 * h + i;
#pragma unroll
    for (int t = 0; t < 4; ++t) {
      AttnOut[((size_t)b * S_LEN + q) * D_DIM + head * 64 + 16 * t + r] =
          f2bf(o[t][i] * inv);
    }
  }
}

// ---------------------------------------------------------------------------
extern "C" void kernel_launch(void* const* d_in, const int* in_sizes, int n_in,
                              void* d_out, int out_size, void* d_ws, size_t ws_size,
                              hipStream_t stream) {
  (void)in_sizes; (void)n_in; (void)out_size; (void)ws_size;

  const float* q    = (const float*)d_in[0];
  const float* k    = (const float*)d_in[1];
  const float* v    = (const float*)d_in[2];
  const int*   mask = (const int*)d_in[3];
  const float* Wq   = (const float*)d_in[4];
  const float* Wk   = (const float*)d_in[5];
  const float* Wv   = (const float*)d_in[6];
  const float* Wo   = (const float*)d_in[7];

  char* ws = (char*)d_ws;
  size_t off = 0;
  const size_t wbytes = (size_t)D_DIM * D_DIM * 2;
  const size_t tbytes = (size_t)M_ROWS * D_DIM * 2;
  unsigned short* Wq_bf = (unsigned short*)(ws + off); off += wbytes;
  unsigned short* Wk_bf = (unsigned short*)(ws + off); off += wbytes;
  unsigned short* Wv_bf = (unsigned short*)(ws + off); off += wbytes;
  unsigned short* Wo_bf = (unsigned short*)(ws + off); off += wbytes;
  unsigned short* Xq_bf = (unsigned short*)(ws + off); off += tbytes;
  unsigned short* Xk_bf = (unsigned short*)(ws + off); off += tbytes;
  unsigned short* Xv_bf = (unsigned short*)(ws + off); off += tbytes;
  unsigned short* Qp    = (unsigned short*)(ws + off); off += tbytes;
  unsigned short* Kp    = (unsigned short*)(ws + off); off += tbytes;
  unsigned short* Vp    = (unsigned short*)(ws + off); off += tbytes;
  unsigned short* Attn  = (unsigned short*)(ws + off); off += tbytes;

  // weight + activation conversion to bf16
  {
    int n8w = D_DIM * D_DIM / 8;
    dim3 cg(n8w / 256), cb(256);
    cvt_f32_to_bf16<<<cg, cb, 0, stream>>>(Wq, Wq_bf, n8w);
    cvt_f32_to_bf16<<<cg, cb, 0, stream>>>(Wk, Wk_bf, n8w);
    cvt_f32_to_bf16<<<cg, cb, 0, stream>>>(Wv, Wv_bf, n8w);
    cvt_f32_to_bf16<<<cg, cb, 0, stream>>>(Wo, Wo_bf, n8w);
    int n8x = M_ROWS * D_DIM / 8;
    dim3 xg(n8x / 256);
    cvt_f32_to_bf16<<<xg, cb, 0, stream>>>(q, Xq_bf, n8x);
    cvt_f32_to_bf16<<<xg, cb, 0, stream>>>(k, Xk_bf, n8x);
    cvt_f32_to_bf16<<<xg, cb, 0, stream>>>(v, Xv_bf, n8x);
  }

  // Q/K/V projections: bf16 in, bf16 head-split out
  dim3 gg(M_ROWS / 128, D_DIM / 64), gb(128);
  gemm_kernel<0><<<gg, gb, 0, stream>>>(Xq_bf, Wq_bf, Qp);
  gemm_kernel<0><<<gg, gb, 0, stream>>>(Xk_bf, Wk_bf, Kp);
  gemm_kernel<0><<<gg, gb, 0, stream>>>(Xv_bf, Wv_bf, Vp);

  // streaming-softmax attention
  dim3 ag(S_LEN / 64, B_SZ * HEADS);
  flash_attn_kernel<<<ag, gb, 0, stream>>>(Qp, Kp, Vp, mask, Attn);

  // output projection (fp32 out)
  gemm_kernel<1><<<gg, gb, 0, stream>>>(Attn, Wo_bf, (float*)d_out);
}
